// RWKV_TimeMix_49340584296473
// MI455X (gfx1250) — compile-verified, matrix operand order checked by reading, not run on hardware
//
#include <hip/hip_runtime.h>
#include <hip/hip_bf16.h>

typedef __bf16 bf16_t;
typedef __attribute__((ext_vector_type(16))) __bf16        v16bf;
typedef __attribute__((ext_vector_type(8)))  float         v8f;
typedef __attribute__((ext_vector_type(4)))  __bf16        bf4;
typedef __attribute__((ext_vector_type(4)))  unsigned int  v4u;
typedef __attribute__((ext_vector_type(8)))  int           v8i;
typedef __attribute__((ext_vector_type(4)))  int           v4i;

#define RW_B 8
#define RW_T 767
#define RW_C 1024
#define RW_BT (RW_B * RW_T)      // 6136
#define RW_MPAD 6144             // 48 * 128

#if __has_builtin(__builtin_amdgcn_tensor_load_to_lds) && \
    __has_builtin(__builtin_amdgcn_s_wait_tensorcnt)
#define HAVE_TDM 1
#else
#define HAVE_TDM 0
#endif

union FragU { uint4 u[2]; v16bf v; };

// ---------------------------------------------------------------------------
// Convert fp32 weights -> bf16 (4 floats per thread)
// ---------------------------------------------------------------------------
__global__ void cvt_bf16_kernel(const float* __restrict__ src,
                                bf16_t* __restrict__ dst, int n4) {
  int i = blockIdx.x * blockDim.x + threadIdx.x;
  if (i < n4) {
    float4 f = ((const float4*)src)[i];
    bf4 o;
    o[0] = (bf16_t)f.x; o[1] = (bf16_t)f.y;
    o[2] = (bf16_t)f.z; o[3] = (bf16_t)f.w;
    ((bf4*)dst)[i] = o;
  }
}

// ---------------------------------------------------------------------------
// xm = x*time_mix + x_prev*(1-time_mix), stored bf16, rows padded to RW_MPAD.
// Pad rows of xm AND rwkv are zeroed so downstream GEMMs read clean zeros.
// ---------------------------------------------------------------------------
__global__ void prep_xm_kernel(const float* __restrict__ x,
                               const float* __restrict__ xx,
                               const float* __restrict__ tmix,
                               bf16_t* __restrict__ xm,
                               bf16_t* __restrict__ rwkv) {
  int idx = blockIdx.x * blockDim.x + threadIdx.x;   // one per 4 elements
  int row = idx >> 8;                                // C/4 = 256 groups per row
  int c4  = (idx & 255) * 4;
  if (row >= RW_MPAD) return;
  if (row < RW_BT) {
    int b = row / RW_T, t = row % RW_T;
    float4 xv = *(const float4*)(x + (size_t)row * RW_C + c4);
    float4 pv = (t == 0) ? *(const float4*)(xx + (size_t)b * RW_C + c4)
                         : *(const float4*)(x + (size_t)(row - 1) * RW_C + c4);
    float4 tv = *(const float4*)(tmix + c4);
    bf4 o;
    o[0] = (bf16_t)(xv.x * tv.x + pv.x * (1.0f - tv.x));
    o[1] = (bf16_t)(xv.y * tv.y + pv.y * (1.0f - tv.y));
    o[2] = (bf16_t)(xv.z * tv.z + pv.z * (1.0f - tv.z));
    o[3] = (bf16_t)(xv.w * tv.w + pv.w * (1.0f - tv.w));
    *(bf4*)(xm + (size_t)row * RW_C + c4) = o;
  } else {
    bf4 z = {};
    *(bf4*)(xm + (size_t)row * RW_C + c4) = z;
    *(bf4*)(rwkv + (size_t)row * RW_C + c4) = z;
  }
}

#if HAVE_TDM
// ---------------------------------------------------------------------------
// TDM: DMA one 128-row x 32-col bf16 tile (row pitch 1024 elems) into LDS.
// 2-D tile: dim0 = K (32, contiguous), dim1 = rows (128). Descriptor per
// CDNA5 ISA 8.3-8.6; groups are block-uniform -> SGPRs. One wave issues,
// completion tracked with TENSORcnt.
// ---------------------------------------------------------------------------
__device__ __forceinline__ void tdm_load_tile_128x32(const bf16_t* gsrc,
                                                     bf16_t* lds_dst,
                                                     unsigned tensor_rows) {
  unsigned long long ga = (unsigned long long)gsrc;
  unsigned lds_off = (unsigned)(unsigned long long)lds_dst;  // AS3 offset

  v4u g0;
  g0[0] = 1u;                                        // count=1, user descriptor
  g0[1] = lds_off;                                   // lds_addr (bytes)
  g0[2] = (unsigned)(ga & 0xFFFFFFFFull);            // global_addr[31:0]
  g0[3] = (unsigned)((ga >> 32) & 0x1FFFFFFull)      // global_addr[56:32]
        | (2u << 30);                                // type = 2 ("image")

  // group1 bit-packing (256 bits, little-endian bit order)
  const unsigned long long DIM0   = 1024;            // tensor extent along K
  const unsigned long long STRIDE = 1024;            // row pitch (elements)
  unsigned long long q0 = (1ull << 16)               // data_size = 1 -> 2 bytes
                        | ((DIM0 & 0xFFFFull) << 48);          // tensor_dim0 lo
  unsigned long long q1 = ((DIM0 >> 16) & 0xFFFFull)           // tensor_dim0 hi
                        | ((unsigned long long)tensor_rows << 16) // tensor_dim1
                        | (32ull << 48);                       // tile_dim0 = 32
  unsigned long long q2 = 128ull                               // tile_dim1 = 128
                        | ((STRIDE & 0xFFFFFFFFull) << 32);    // dim0_stride lo
  unsigned long long q3 = (STRIDE >> 32) & 0xFFFFull;          // dim0_stride hi

  v8i g1;
  g1[0] = (int)q0; g1[1] = (int)(q0 >> 32);
  g1[2] = (int)q1; g1[3] = (int)(q1 >> 32);
  g1[4] = (int)q2; g1[5] = (int)(q2 >> 32);
  g1[6] = (int)q3; g1[7] = (int)(q3 >> 32);

  v4i g2 = {0, 0, 0, 0};
  v4i g3 = {0, 0, 0, 0};
#if defined(__clang_major__) && (__clang_major__ >= 23)
  v8i g4 = {0, 0, 0, 0, 0, 0, 0, 0};   // amdgpu-toolchain 6-arg form
  __builtin_amdgcn_tensor_load_to_lds(g0, g1, g2, g3, g4, 0);
#else
  __builtin_amdgcn_tensor_load_to_lds(g0, g1, g2, g3, 0);     // ROCm 7.2 form
#endif
}
#endif  // HAVE_TDM

// ---------------------------------------------------------------------------
// O[m,n] = sum_k A[m,k] * W[n,k]   (A: MPADxK bf16, W: NxK bf16, O: f32)
// Block tile 128x128, 256 threads (8 waves). Wave tile 32x64 = 2x4 WMMA.
// Double-buffered LDS; tiles fetched by the Tensor Data Mover (wave 0 issues,
// s_wait_tensorcnt + barrier publishes), WMMA overlaps the next tile's DMA.
// ---------------------------------------------------------------------------
__global__ __launch_bounds__(256, 2)
void gemm_bf16_wmma_kernel(const bf16_t* __restrict__ A,
                           const bf16_t* __restrict__ W,
                           float* __restrict__ O,
                           int Mvalid, int Arows) {
  const int K = RW_C, N = RW_C;
  __shared__ bf16_t As[2][128 * 32];
  __shared__ bf16_t Bs[2][128 * 32];

  const int tid  = threadIdx.x;
  const int lane = tid & 31;
  const int wave = tid >> 5;
  const int wm   = wave & 3;        // 4 waves along M (32 rows each)
  const int wn   = wave >> 2;       // 2 waves along N (64 cols each)
  const int half = lane >> 4;
  const int lr   = lane & 15;
  const int m0   = blockIdx.x * 128;
  const int n0   = blockIdx.y * 128;

  v8f acc[2][4];
#pragma unroll
  for (int sm = 0; sm < 2; ++sm)
#pragma unroll
    for (int sn = 0; sn < 4; ++sn) {
      v8f z = {0.f, 0.f, 0.f, 0.f, 0.f, 0.f, 0.f, 0.f};
      acc[sm][sn] = z;
    }

  const int KT = K / 32;

#if HAVE_TDM
  if (wave == 0) {   // prologue: DMA tile 0 into buffer 0
    tdm_load_tile_128x32(A + (size_t)m0 * K, &As[0][0], (unsigned)Arows);
    tdm_load_tile_128x32(W + (size_t)n0 * K, &Bs[0][0], (unsigned)N);
  }
#endif

  for (int kt = 0; kt < KT; ++kt) {
    const int cur = kt & 1;
#if HAVE_TDM
    if (wave == 0) __builtin_amdgcn_s_wait_tensorcnt(0);  // cur tiles in LDS
    __syncthreads();                                      // publish to all waves
    if (wave == 0 && (kt + 1) < KT) {                     // overlap next DMA
      tdm_load_tile_128x32(A + (size_t)m0 * K + (kt + 1) * 32,
                           &As[cur ^ 1][0], (unsigned)Arows);
      tdm_load_tile_128x32(W + (size_t)n0 * K + (kt + 1) * 32,
                           &Bs[cur ^ 1][0], (unsigned)N);
    }
#else
    __syncthreads();
#pragma unroll
    for (int rep = 0; rep < 2; ++rep) {
      int thr = tid + rep * 256;
      int row = thr >> 2, seg = thr & 3;     // 128 rows x 4 x 16B = 8KB tile
      *(uint4*)(&As[cur][row * 32 + seg * 8]) =
          *(const uint4*)(A + (size_t)(m0 + row) * K + kt * 32 + seg * 8);
      *(uint4*)(&Bs[cur][row * 32 + seg * 8]) =
          *(const uint4*)(W + (size_t)(n0 + row) * K + kt * 32 + seg * 8);
    }
    if (kt + 1 < KT) {
      __builtin_prefetch(A + (size_t)(m0 + (tid >> 1)) * K + (kt + 1) * 32, 0, 0);
      __builtin_prefetch(W + (size_t)(n0 + (tid >> 1)) * K + (kt + 1) * 32, 0, 0);
    }
    __syncthreads();
#endif

    v16bf af[2], bfr[4];
#pragma unroll
    for (int sm = 0; sm < 2; ++sm) {
      int row = wm * 32 + sm * 16 + lr;
      FragU fu;
      fu.u[0] = *(const uint4*)(&As[cur][row * 32 + half * 8]);
      fu.u[1] = *(const uint4*)(&As[cur][row * 32 + 16 + half * 8]);
      af[sm] = fu.v;
    }
#pragma unroll
    for (int sn = 0; sn < 4; ++sn) {
      int row = wn * 64 + sn * 16 + lr;
      FragU fu;
      fu.u[0] = *(const uint4*)(&Bs[cur][row * 32 + half * 8]);
      fu.u[1] = *(const uint4*)(&Bs[cur][row * 32 + 16 + half * 8]);
      bfr[sn] = fu.v;
    }
#pragma unroll
    for (int sm = 0; sm < 2; ++sm)
#pragma unroll
      for (int sn = 0; sn < 4; ++sn)
        acc[sm][sn] = __builtin_amdgcn_wmma_f32_16x16x32_bf16(
            false, af[sm], false, bfr[sn], (short)0, acc[sm][sn], false, false);
  }

  // C/D layout: VGPR r, lane l -> M = r + 8*(l>>4), N = l&15
#pragma unroll
  for (int sm = 0; sm < 2; ++sm) {
    int rbase = m0 + wm * 32 + sm * 16 + half * 8;
#pragma unroll
    for (int sn = 0; sn < 4; ++sn) {
      int col = n0 + wn * 64 + sn * 16 + lr;
#pragma unroll
      for (int r = 0; r < 8; ++r) {
        int row = rbase + r;
        if (row < Mvalid) O[(size_t)row * N + col] = acc[sm][sn][r];
      }
    }
  }
}

// ---------------------------------------------------------------------------
// WKV: the causal grouped conv with exp-decay kernel == linear recurrence.
// One thread per (b, d); coalesced across d at every timestep.
// ---------------------------------------------------------------------------
__global__ void wkv_scan_kernel(const float* __restrict__ kf,
                                const float* __restrict__ vf,
                                const float* __restrict__ rf,
                                const float* __restrict__ aa,
                                const float* __restrict__ bb,
                                const float* __restrict__ mmp,
                                const float* __restrict__ td,
                                const float* __restrict__ tfp,
                                bf16_t* __restrict__ rwkv) {
  int d = blockIdx.x * blockDim.x + threadIdx.x;   // 0..C-1
  int b = blockIdx.y;                              // 0..B-1
  size_t base = (size_t)b * RW_T * RW_C + d;

  // pass 1: m = max_t k[b,d,t]
  float m = -3.4028235e38f;
  for (int t = 0; t < RW_T; ++t)
    m = fmaxf(m, kf[base + (size_t)t * RW_C]);

  // mm_new = ldexp(0.5, frexp_exp(m))
  int e;
  (void)frexpf(m, &e);
  float mmn = ldexpf(0.5f, e);

  float rsf = expf(mmp[b * RW_C + d] - mmn);
  float tfv = expf(tfp[d]);
  float w1  = expf(-expf(td[d]));

  float S   = bb[b * RW_C + d] * rsf;   // wk state
  float Acc = aa[b * RW_C + d] * rsf;   // wkv state

  for (int t = 0; t < RW_T; ++t) {
    size_t off = base + (size_t)t * RW_C;
    float kk = kf[off];
    float vv = vf[off];
    float ke = expf(kk - mmn);
    float kv = ke * vv;
    float wk  = tfv * ke + S;
    float wkv = tfv * kv + Acc;
    float q = wkv / wk;
    if (q != q) q = 0.0f;                                // nan_to_num: NaN -> 0
    else if (q >  3.4028235e38f) q =  3.4028235e38f;     // +inf -> max
    else if (q < -3.4028235e38f) q = -3.4028235e38f;     // -inf -> min
    float rr  = rf[off];
    float sig = 1.0f / (1.0f + expf(-rr));
    rwkv[off] = (bf16_t)(sig * q);
    S   = w1 * S + ke;
    Acc = w1 * Acc + kv;
  }
}

// ---------------------------------------------------------------------------
extern "C" void kernel_launch(void* const* d_in, const int* in_sizes, int n_in,
                              void* d_out, int out_size, void* d_ws, size_t ws_size,
                              hipStream_t stream) {
  const float* x    = (const float*)d_in[0];
  const float* xx   = (const float*)d_in[1];
  const float* aa   = (const float*)d_in[2];
  const float* bb   = (const float*)d_in[3];
  const float* mm   = (const float*)d_in[4];
  const float* td   = (const float*)d_in[5];   // time_decay (C)
  const float* tf   = (const float*)d_in[6];   // time_first (C)
  const float* tmix = (const float*)d_in[7];   // time_mix   (C)
  const float* Wk   = (const float*)d_in[8];
  const float* Wv   = (const float*)d_in[9];
  const float* Wr   = (const float*)d_in[10];
  const float* Wo   = (const float*)d_in[11];
  float* out = (float*)d_out;

  // workspace carve-out (256B aligned)
  size_t off = 0;
  auto carve = [&](size_t bytes) -> char* {
    char* p = (char*)d_ws + off;
    off = (off + bytes + 255) & ~(size_t)255;
    return p;
  };
  bf16_t* xm_bf   = (bf16_t*)carve((size_t)RW_MPAD * RW_C * sizeof(bf16_t));
  bf16_t* Wk_bf   = (bf16_t*)carve((size_t)RW_C * RW_C * sizeof(bf16_t));
  bf16_t* Wv_bf   = (bf16_t*)carve((size_t)RW_C * RW_C * sizeof(bf16_t));
  bf16_t* Wr_bf   = (bf16_t*)carve((size_t)RW_C * RW_C * sizeof(bf16_t));
  bf16_t* Wo_bf   = (bf16_t*)carve((size_t)RW_C * RW_C * sizeof(bf16_t));
  float*  k_f     = (float*)carve((size_t)RW_BT * RW_C * sizeof(float));
  float*  v_f     = (float*)carve((size_t)RW_BT * RW_C * sizeof(float));
  float*  r_f     = (float*)carve((size_t)RW_BT * RW_C * sizeof(float));
  bf16_t* rwkv_bf = (bf16_t*)carve((size_t)RW_MPAD * RW_C * sizeof(bf16_t));

  // 1) weights -> bf16
  {
    int n4 = RW_C * RW_C / 4;
    dim3 g((n4 + 255) / 256);
    cvt_bf16_kernel<<<g, 256, 0, stream>>>(Wk, Wk_bf, n4);
    cvt_bf16_kernel<<<g, 256, 0, stream>>>(Wv, Wv_bf, n4);
    cvt_bf16_kernel<<<g, 256, 0, stream>>>(Wr, Wr_bf, n4);
    cvt_bf16_kernel<<<g, 256, 0, stream>>>(Wo, Wo_bf, n4);
  }

  // 2) time-mix + pad
  prep_xm_kernel<<<dim3(RW_MPAD), 256, 0, stream>>>(x, xx, tmix, xm_bf, rwkv_bf);

  // 3) k, v, r projections (WMMA bf16, fp32 accumulate, TDM-fed)
  dim3 gg(RW_MPAD / 128, RW_C / 128);
  gemm_bf16_wmma_kernel<<<gg, 256, 0, stream>>>(xm_bf, Wk_bf, k_f, RW_BT, RW_MPAD);
  gemm_bf16_wmma_kernel<<<gg, 256, 0, stream>>>(xm_bf, Wv_bf, v_f, RW_BT, RW_MPAD);
  gemm_bf16_wmma_kernel<<<gg, 256, 0, stream>>>(xm_bf, Wr_bf, r_f, RW_BT, RW_MPAD);

  // 4) WKV recurrence (replaces O(T^2) conv) + sigmoid(r) gating
  wkv_scan_kernel<<<dim3(RW_C / 256, RW_B), 256, 0, stream>>>(
      k_f, v_f, r_f, aa, bb, mm, td, tf, rwkv_bf);

  // 5) output projection straight into d_out
  gemm_bf16_wmma_kernel<<<gg, 256, 0, stream>>>(rwkv_bf, Wo_bf, out, RW_BT, RW_MPAD);
}
